// FilteredNoiseGenerator_15899968929905
// MI455X (gfx1250) — compile-verified, matrix operand order checked by reading, not run on hardware
//
#include <hip/hip_runtime.h>

typedef __attribute__((ext_vector_type(16))) _Float16 v16h;
typedef __attribute__((ext_vector_type(8)))  float    v8f;

#define FS      80                    // framesize (hop)
#define NB      257                   // nbands
#define NFIR    513                   // FIR length N = 2*NB-1
#define NFFT    (FS + NFIR - 1)       // 592 = full-conv length per frame
#define KPAD    288                   // 257 padded to multiple of 32
#define KT      (KPAD / 32)           // 9 WMMA K-steps
#define NPAD    528                   // 513 padded to multiple of 16
#define NT      (NPAD / 16)           // 33 WMMA N-tiles
#define FPB     16                    // frames per block (WMMA M tile)
#define WIN     (FPB * FS + NFIR - 1) // 1792 = OLA window per tile
#define TFRAMES 4000                  // frames per batch (reference t)
#define SPB     (TFRAMES * FS)        // 320000 output samples per batch

// 16-bit WMMA fragment K permutation (ISA 7.12.2, "16-bit A-Matrix 16x32"):
// lanes 0-15 hold K = {0..7, 16..23}; lanes 16-31 hold K = {8..15, 24..31}.
__device__ __forceinline__ int kmap(int lane, int e) {
    int hi   = (lane >> 4) & 1;
    int base = (e < 8) ? (hi ? 8 : 0) : (hi ? 24 : 16);
    return base + (e & 7);
}

// Precompute the irFFT+roll+Hann basis C[k][n] in f16, stored directly in the
// per-lane WMMA B-fragment order: flat index = ((kt*NT + nt)*32 + lane)*16 + e.
__global__ __launch_bounds__(256) void build_basis(_Float16* __restrict__ basis) {
    int p = blockIdx.x * 256 + threadIdx.x;
    const int total = KT * NT * 32 * 16;
    if (p >= total) return;
    int e    = p & 15;
    int lane = (p >> 4) & 31;
    int nt   = (p >> 9) % NT;
    int kt   = (p >> 9) / NT;
    int k = kt * 32 + kmap(lane, e);
    int n = nt * 16 + (lane & 15);
    float v = 0.f;
    if (k < NB && n < NFIR) {
        const float invN = 1.0f / (float)NFIR;
        const float twoPi = 6.283185307179586f;
        float hann = 0.5f * (1.0f - cosf(twoPi * (float)n * invN));     // periodic Hann
        float wk   = (k == 0 ? 1.0f : 2.0f) * invN;                     // irfft weights
        v = hann * wk * cosf(twoPi * (float)k * (float)(n - (NB - 1)) * invN);
    }
    basis[p] = (_Float16)v;
}

__global__ __launch_bounds__(256) void zero_kernel(float* __restrict__ p, long n) {
    long i = (long)blockIdx.x * blockDim.x + threadIdx.x;
    if (i < n) p[i] = 0.f;
}

__global__ __launch_bounds__(256) void ltv_fir_kernel(
    const float* __restrict__ H, const float* __restrict__ noise_u,
    const _Float16* __restrict__ basis, float* __restrict__ out,
    long total_frames) {
    __shared__ float hRow[FPB][NB + 3];                 // staged H tile (coalesced load)
    __shared__ __align__(32) _Float16 aFrag[KT][32][16];// A fragments (f16, lane layout)
    __shared__ float fir[FPB][NPAD];                    // GEMM result: firwin per frame
    __shared__ float nz[FPB][FS];                       // noise tile mapped to [-1,1)

    const int tid  = threadIdx.x;
    const int lane = tid & 31;
    const int wave = tid >> 5;                          // 8 wave32's per block
    const long frame0 = (long)blockIdx.x * FPB;

    // Stage 0: coalesced load of H rows + noise tile
    for (int p = tid; p < FPB * NB; p += 256) {
        int f = p / NB, k = p - f * NB;
        long g = frame0 + f;
        hRow[f][k] = (g < total_frames) ? H[g * NB + k] : 0.f;
    }
    for (int p = tid; p < FPB * FS; p += 256) {
        int f = p / FS, i = p - f * FS;
        long g = frame0 + f;
        nz[f][i] = (g < total_frames) ? noise_u[g * FS + i] * 2.0f - 1.0f : 0.f;
    }
    __syncthreads();

    // Stage 1: repack H tile into per-lane A fragments (f16, zero-padded K)
    for (int p = tid; p < KT * 32 * 16; p += 256) {
        int e  = p & 15;
        int l  = (p >> 4) & 31;
        int kt = p >> 9;
        int row = l & 15;
        int k = kt * 32 + kmap(l, e);
        aFrag[kt][l][e] = (_Float16)((k < NB) ? hRow[row][k] : 0.f);
    }
    __syncthreads();

    // Stage 2: firwin[16][513] = H_tile x C via v_wmma_f32_16x16x32_f16
    for (int nt = wave; nt < NT; nt += 8) {
        v8f acc = {};
        const _Float16* bcol = basis + ((long)nt * 32 + lane) * 16;
#pragma unroll
        for (int kt = 0; kt < KT; ++kt) {
            v16h a = *(const v16h*)&aFrag[kt][lane][0];
            v16h b = *(const v16h*)&bcol[(long)kt * NT * 32 * 16];
            acc = __builtin_amdgcn_wmma_f32_16x16x32_f16(
                /*neg_a=*/false, a, /*neg_b=*/false, b,
                /*c_mod=*/(short)0, acc, /*reuse_a=*/false, /*reuse_b=*/false);
        }
        // D layout (ISA 7.12.2): element r -> (M = r + 8*(lane>=16), N = lane%16)
        int n0  = nt * 16 + (lane & 15);
        int mhi = (lane >> 4) << 3;
#pragma unroll
        for (int r = 0; r < 8; ++r)
            fir[r + mhi][n0] = acc[r];
    }
    __syncthreads();

    // Stage 3: fused full convolution + intra-tile overlap-add.
    // Each thread owns distinct window positions -> register accumulate,
    // single global atomic per sample (inter-tile overlap via neighbor blocks).
    const long b0 = frame0 / TFRAMES;       // tiles never straddle batches (4000 % 16 == 0)
    const long t0 = frame0 - b0 * TFRAMES;
    float* outb = out + b0 * (long)SPB;
    for (int w = tid; w < WIN; w += 256) {
        float s = 0.f;
        int fmin = (w - (NFFT - 1) <= 0) ? 0 : (w - (NFFT - 1) + FS - 1) / FS;
        int fmax = w / FS; if (fmax > FPB - 1) fmax = FPB - 1;
        for (int f = fmin; f <= fmax; ++f) {
            int j  = w - f * FS;                       // 0..591 within frame f's conv
            int i0 = j - (NFIR - 1); if (i0 < 0) i0 = 0;
            int i1 = (j < FS - 1) ? j : FS - 1;
            const float* fr = &fir[f][0];
            const float* nr = &nz[f][0];
            for (int i = i0; i <= i1; ++i) s += nr[i] * fr[j - i];
        }
        long pos = t0 * FS + w;
        if (pos < SPB)                                  // tail trim, as in reference
            atomicAdd(&outb[pos], s);
    }
}

extern "C" void kernel_launch(void* const* d_in, const int* in_sizes, int n_in,
                              void* d_out, int out_size, void* d_ws, size_t ws_size,
                              hipStream_t stream) {
    const float* H       = (const float*)d_in[0];   // [B, t, 257] f32
    const float* noise_u = (const float*)d_in[1];   // [B, t, 80]  f32
    float* out = (float*)d_out;                     // [B, t*80]   f32
    _Float16* basis = (_Float16*)d_ws;              // ~297 KB f16 basis in scratch

    long total_frames = (long)in_sizes[1] / FS;     // B*t = 128000

    const int basis_elems = KT * NT * 32 * 16;
    build_basis<<<(basis_elems + 255) / 256, 256, 0, stream>>>(basis);
    zero_kernel<<<(int)(((long)out_size + 255) / 256), 256, 0, stream>>>(out, (long)out_size);
    int blocks = (int)((total_frames + FPB - 1) / FPB);
    ltv_fir_kernel<<<blocks, 256, 0, stream>>>(H, noise_u, basis, out, total_frames);
}
